// CondConvBNReLUV2_70222715289955
// MI455X (gfx1250) — matrix operand order repaired and checked by reading, not compiled, not run on hardware
//
#include <hip/hip_runtime.h>
#include <hip/hip_bf16.h>

typedef __attribute__((ext_vector_type(16))) __bf16        v16bf;
typedef __attribute__((ext_vector_type(8)))  float          v8f;
typedef __attribute__((ext_vector_type(16))) unsigned short v16u;

#define B_    32
#define E_    8
#define CIN_  64
#define COUT_ 64
#define HH_   112
#define WW_   112
#define HW_   12544            // 112*112
#define KTOT_ 576              // Cin*3*3
#define TN_   128              // pixel tile per block
#define NTILE_ 98              // HW_/TN_
#define TOTAL_OUT_ 25690112    // B*Cout*HW
#define NSTAT_ 401408.0f       // B*HW (BN population per channel)

union FragU { uint4 q[2]; v16u u; v16bf b; };

__device__ __forceinline__ unsigned short f2bf(float f) {
    union { float f; unsigned u; } v; v.f = f;
    unsigned r = v.u + 0x7FFFu + ((v.u >> 16) & 1u);   // RNE
    return (unsigned short)(r >> 16);
}

// ---------------------------------------------------------------------------
// Kernel 1: per-sample expert mix -> bf16 weights [B][Cout][576]; zero stats.
// ---------------------------------------------------------------------------
__global__ __launch_bounds__(256)
void k_combine(const float* __restrict__ rw,        // [B][E]
               const float* __restrict__ experts,   // [E][Cout][576]
               unsigned short* __restrict__ wk,     // [B][Cout][576] bf16
               float* __restrict__ stats)           // 128 floats (sum|sumsq)
{
    unsigned i = blockIdx.x * 256u + threadIdx.x;    // 4608 blocks exactly
    if (i < 128u) stats[i] = 0.0f;
    if (i >= (unsigned)(B_ * COUT_ * KTOT_)) return;
    int k = i % KTOT_;
    int o = (i / KTOT_) % COUT_;
    int b = i / (KTOT_ * COUT_);
    float s = 0.0f;
    #pragma unroll
    for (int e = 0; e < E_; ++e)
        s += rw[b * E_ + e] * experts[(e * COUT_ + o) * KTOT_ + k];
    wk[i] = f2bf(s);
}

// ---------------------------------------------------------------------------
// Kernel 2: implicit-GEMM conv via v_wmma_f32_16x16x32_bf16.
//   Block: 256 thr = 8 waves. Tile: M=64 (Cout) x N=128 pixels, K step 32.
//   As: row-major [Cout][32].  Bs: pixel-major [128][32] so each lane's
//   16 K-values form two runs of 8 (k0.., 16+k0..) -> 2x ds_load_b128 per fragment.
// ---------------------------------------------------------------------------
__global__ __launch_bounds__(256)
void k_conv_wmma(const float* __restrict__ x,           // [B][Cin][112][112]
                 const unsigned short* __restrict__ wk, // [B][Cout][576] bf16
                 float* __restrict__ out)               // [B][Cout][HW]
{
    __shared__ unsigned short As[COUT_ * 32];   // 64x32 bf16  (4 KB)
    __shared__ unsigned short Bs[TN_ * 32];     // 128x32 bf16 (8 KB)

    const int blk  = blockIdx.x;
    const int b    = blk / NTILE_;
    const int p0   = (blk % NTILE_) * TN_;
    const int t    = threadIdx.x;
    const int lane = t & 31;
    const int wave = t >> 5;

    const float*          xb = x  + (size_t)b * CIN_ * HW_;
    const unsigned short* wb = wk + (size_t)b * COUT_ * KTOT_;

    // per-thread staging geometry (fixed across K loop)
    const int aidx = t * 8;                  // A stage: 8 bf16 per thread
    const int arow_s = aidx >> 5;
    const int acol_s = aidx & 31;
    const int pix   = t >> 1;                // B stage: one pixel, 16 K-rows
    const int khalf = (t & 1) << 4;          // 0 or 16
    const int p     = p0 + pix;
    const int h     = p / WW_;
    const int w     = p - h * WW_;

    v8f acc[4];
    #pragma unroll
    for (int m = 0; m < 4; ++m) acc[m] = (v8f)0.0f;

    for (int kk = 0; kk < KTOT_; kk += 32) {
        // ---- stage A: 64x32 bf16, one 16B vector copy per thread ----
        {
            const uint4* src = reinterpret_cast<const uint4*>(wb + arow_s * KTOT_ + kk + acol_s);
            *reinterpret_cast<uint4*>(&As[aidx]) = *src;
            if (kk + 32 < KTOT_)             // gfx1250 global_prefetch_b8
                __builtin_prefetch(wb + arow_s * KTOT_ + kk + 32 + acol_s, 0, 3);
        }
        // ---- stage B: im2col, pixel-major; one h/w decompose per thread ----
        {
            int kidx = kk + khalf;
            int ci   = kidx / 9;
            int rem  = kidx - ci * 9;
            int kh   = rem / 3;
            int kw   = rem - kh * 3;
            FragU st;
            #pragma unroll
            for (int j = 0; j < 16; ++j) {
                int ih = h + kh - 1;
                int iw = w + kw - 1;
                float v = 0.0f;
                if ((unsigned)ih < (unsigned)HH_ && (unsigned)iw < (unsigned)WW_)
                    v = xb[ci * HW_ + ih * WW_ + iw];
                st.u[j] = f2bf(v);
                ++kw;
                if (kw == 3) { kw = 0; ++kh; if (kh == 3) { kh = 0; ++ci; } }
            }
            uint4* dst = reinterpret_cast<uint4*>(&Bs[pix * 32 + khalf]);
            dst[0] = st.q[0];
            dst[1] = st.q[1];
        }
        __syncthreads();

        // ---- B fragment: lane n = wave*16 + (lane&15); K k0..k0+7 / 16+k0.. ----
        const int nloc = wave * 16 + (lane & 15);
        const int kofB = (lane >> 4) * 8;           // lanes16-31: K 8..15 / 24..31
        FragU bf;
        {
            const unsigned short* bsrc = &Bs[nloc * 32 + kofB];
            bf.q[0] = *reinterpret_cast<const uint4*>(bsrc);
            bf.q[1] = *reinterpret_cast<const uint4*>(bsrc + 16);
        }

        // ---- A fragments (2x b128 each) + 4 WMMAs ----
        const int arow = lane & 15;
        const int k0   = (lane >> 4) * 8;           // lanes16-31: K 8..15 / 24..31
        #pragma unroll
        for (int m = 0; m < 4; ++m) {
            const unsigned short* ar = &As[(m * 16 + arow) * 32];
            FragU af;
            af.q[0] = *reinterpret_cast<const uint4*>(ar + k0);        // K k0..k0+7
            af.q[1] = *reinterpret_cast<const uint4*>(ar + 16 + k0);   // K 16+k0..
            acc[m] = __builtin_amdgcn_wmma_f32_16x16x32_bf16(
                false, af.b, false, bf.b, (short)0, acc[m], false, false);
        }
        __syncthreads();
    }

    // ---- write back (C/D f32 layout: VGPR r -> M=r / M=8+r per half-wave) ----
    const int nloc = wave * 16 + (lane & 15);
    const int pw   = p0 + nloc;
    const int mo   = (lane >> 4) * 8;
    #pragma unroll
    for (int m = 0; m < 4; ++m) {
        #pragma unroll
        for (int r = 0; r < 8; ++r) {
            int M = m * 16 + mo + r;
            out[((size_t)b * COUT_ + M) * HW_ + pw] = acc[m][r];
        }
    }
}

// ---------------------------------------------------------------------------
// Kernel 3: per-channel sum / sumsq. One block per (b,c) plane.
// ---------------------------------------------------------------------------
__global__ __launch_bounds__(256)
void k_reduce(const float* __restrict__ out, float* __restrict__ stats)
{
    __shared__ float sh[256], sh2[256];
    const int plane = blockIdx.x;          // b*64 + c  (2048 blocks)
    const int c     = plane & 63;
    const float* p  = out + (size_t)plane * HW_;
    float s = 0.0f, ss = 0.0f;
    for (int i = threadIdx.x; i < HW_; i += 256) {
        float v = p[i];
        s  += v;
        ss += v * v;
    }
    sh[threadIdx.x]  = s;
    sh2[threadIdx.x] = ss;
    __syncthreads();
    for (int st = 128; st > 0; st >>= 1) {
        if (threadIdx.x < st) {
            sh[threadIdx.x]  += sh[threadIdx.x + st];
            sh2[threadIdx.x] += sh2[threadIdx.x + st];
        }
        __syncthreads();
    }
    if (threadIdx.x == 0) {
        atomicAdd(&stats[c],      sh[0]);
        atomicAdd(&stats[64 + c], sh2[0]);
    }
}

// ---------------------------------------------------------------------------
// Kernel 4: BN (batch stats) + ReLU6, in place on d_out.
// ---------------------------------------------------------------------------
__global__ __launch_bounds__(256)
void k_bn_relu6(float* __restrict__ out,
                const float* __restrict__ stats,
                const float* __restrict__ gamma,
                const float* __restrict__ beta)
{
    size_t i = (size_t)blockIdx.x * 256u + threadIdx.x;   // 100352 blocks exact
    if (i >= (size_t)TOTAL_OUT_) return;
    int c = (int)((i / HW_) & 63);
    const float invN = 1.0f / NSTAT_;
    float mean = stats[c] * invN;
    float var  = stats[64 + c] * invN - mean * mean;
    float v = (out[i] - mean) * rsqrtf(var + 1e-5f) * gamma[c] + beta[c];
    out[i] = fminf(fmaxf(v, 0.0f), 6.0f);
}

// ---------------------------------------------------------------------------
extern "C" void kernel_launch(void* const* d_in, const int* in_sizes, int n_in,
                              void* d_out, int out_size, void* d_ws, size_t ws_size,
                              hipStream_t stream)
{
    const float* x       = (const float*)d_in[0];   // [32][64][112][112]
    const float* rw      = (const float*)d_in[1];   // [32][8]
    const float* experts = (const float*)d_in[2];   // [8][64][64][3][3]
    const float* gamma   = (const float*)d_in[3];   // [64]
    const float* beta    = (const float*)d_in[4];   // [64]
    float* out           = (float*)d_out;           // [32][64][112][112]

    // workspace layout: [0, 2359296) bf16 weights ; then 128 f32 stats
    unsigned short* wk = (unsigned short*)d_ws;
    float* stats = (float*)((char*)d_ws + (size_t)B_ * COUT_ * KTOT_ * sizeof(unsigned short));

    k_combine<<<(B_ * COUT_ * KTOT_) / 256, 256, 0, stream>>>(rw, experts, wk, stats);
    k_conv_wmma<<<B_ * NTILE_, 256, 0, stream>>>(x, wk, out);
    k_reduce<<<B_ * COUT_, 256, 0, stream>>>(out, stats);
    k_bn_relu6<<<TOTAL_OUT_ / 256, 256, 0, stream>>>(out, stats, gamma, beta);
}
